// MeanFieldInference_54528904790122
// MI455X (gfx1250) — compile-verified, matrix operand order checked by reading, not run on hardware
//
#include <hip/hip_runtime.h>
#include <hip/hip_bf16.h>

#define D_EDGE 16
#define N_GRAPHS 64

typedef __attribute__((ext_vector_type(16))) __bf16 v16bf;
typedef __attribute__((ext_vector_type(8)))  __bf16 v8bf;
typedef __attribute__((ext_vector_type(4)))  __bf16 v4bf;
typedef __attribute__((ext_vector_type(8)))  float  v8f;

union frag16 { v16bf v; v8bf h[2]; };

// ---------------------------------------------------------------------------
// f32 -> bf16 elementwise conversion (native casts -> packed v_cvt on gfx1250).
__global__ void k_cvt_bf16(const float* __restrict__ in, __bf16* __restrict__ out, int n4) {
    int t = blockIdx.x * blockDim.x + threadIdx.x;
    if (t >= n4) return;
    float4 f = ((const float4*)in)[t];
    v4bf o;
    o[0] = (__bf16)f.x; o[1] = (__bf16)f.y; o[2] = (__bf16)f.z; o[3] = (__bf16)f.w;
    ((v4bf*)out)[t] = o;
}

// ---------------------------------------------------------------------------
// D = A(bf16)[rows,128] @ W(bf16)[128 out,128 in]^T, compile-time bias+relu.
// One wave per 16-wide column tile; 8 waves/block cover all 128 output cols.
// Each wave processes TILES_PER_WAVE row-tiles, holding its B fragments
// (its 16 columns of W across all K) in registers.
//
// Fragment layout per CDNA5 ISA 7.12.2 (16-bit A 16x32):
//   lane l in [0,15]  : row M=l,    K = {kb+0..7},  {kb+16..23}
//   lane l in [16,31] : row M=l-16, K = {kb+8..15}, {kb+24..31}
// B symmetric over columns (B[k][n] = W[c0+n][k], contiguous row of W).
// C/D f32 16x16: lane l, elem v -> row M = v + 8*(l>=16), col N = l%16,
// so a lane's 8 outputs sit 128 floats apart in one column: immediate-offset
// stores off a single base pointer.
#define TILES_PER_WAVE 5
template <bool HAS_BIAS, bool DO_RELU>
__global__ void k_gemm_wmma(const __bf16* __restrict__ A,
                            const __bf16* __restrict__ W,
                            const float*  __restrict__ bias,
                            float* __restrict__ D,
                            int nRows) {
    const int lane = threadIdx.x & 31;
    const int wave = threadIdx.x >> 5;          // 0..7 -> column tile
    const int c0   = wave * 16;
    const int half = lane >> 4;
    const int m    = lane & 15;
    const int rBase = blockIdx.x * (TILES_PER_WAVE * 16);

    // Load B fragments once: 4 K-blocks of this wave's 16 columns.
    const __bf16* brow = W + (size_t)(c0 + m) * 128;
    frag16 bf[4];
#pragma unroll
    for (int kb = 0; kb < 4; ++kb) {
        bf[kb].h[0] = *(const v8bf*)(brow + kb * 32 + 8 * half);
        bf[kb].h[1] = *(const v8bf*)(brow + kb * 32 + 16 + 8 * half);
    }

#pragma unroll
    for (int t = 0; t < TILES_PER_WAVE; ++t) {
        const int r0 = rBase + t * 16;
        if (r0 >= nRows) break;                  // wave-uniform
        const __bf16* arow = A + (size_t)(r0 + m) * 128;

        v8f acc = {};
#pragma unroll
        for (int kb = 0; kb < 4; ++kb) {
            frag16 af;
            af.h[0] = *(const v8bf*)(arow + kb * 32 + 8 * half);
            af.h[1] = *(const v8bf*)(arow + kb * 32 + 16 + 8 * half);
            acc = __builtin_amdgcn_wmma_f32_16x16x32_bf16(
                false, af.v, false, bf[kb].v, (short)0, acc, false, false);
        }

        const size_t base = (size_t)(r0 + 8 * half) * 128 + (size_t)(c0 + m);
        float* dp = D + base;
        if (HAS_BIAS) {
            const float* bp = bias + base;
            float bv[8];
#pragma unroll
            for (int v = 0; v < 8; ++v) bv[v] = bp[v * 128];
#pragma unroll
            for (int v = 0; v < 8; ++v) {
                float val = acc[v] + bv[v];
                if (DO_RELU) val = fmaxf(val, 0.0f);
                dp[v * 128] = val;
            }
        } else {
#pragma unroll
            for (int v = 0; v < 8; ++v) {
                float val = acc[v];
                if (DO_RELU) val = fmaxf(val, 0.0f);
                dp[v * 128] = val;
            }
        }
    }
}

// ---------------------------------------------------------------------------
// Per-edge: emb = edge_feat[e] (16) @ W_e2l^T (128x16), atomicAdd into
// input_message[edge_dst[e]]. One wave/edge, 4 outputs/lane, W in LDS
// (transposed: Ws[k*128+o]) to spread banks.
__global__ void k_edge_emb(const float* __restrict__ edge_feat,
                           const int* __restrict__ edge_dst,
                           const float* __restrict__ W_e2l,
                           float* __restrict__ im, int nEdges) {
    __shared__ float Ws[D_EDGE * 128];
    for (int i = threadIdx.x; i < D_EDGE * 128; i += blockDim.x) {
        int o = i >> 4, k = i & 15;
        Ws[k * 128 + o] = W_e2l[i];
    }
    __syncthreads();

    int t = blockIdx.x * blockDim.x + threadIdx.x;
    int e = t >> 5, lane = t & 31;
    if (e >= nEdges) return;

    const float* f = edge_feat + (size_t)e * D_EDGE;
    float fk[D_EDGE];
#pragma unroll
    for (int k = 0; k < D_EDGE; ++k) fk[k] = f[k];

    const int o0 = lane * 4;
    float acc0 = 0.f, acc1 = 0.f, acc2 = 0.f, acc3 = 0.f;
#pragma unroll
    for (int k = 0; k < D_EDGE; ++k) {
        const float* w = &Ws[k * 128 + o0];
        acc0 = fmaf(fk[k], w[0], acc0);
        acc1 = fmaf(fk[k], w[1], acc1);
        acc2 = fmaf(fk[k], w[2], acc2);
        acc3 = fmaf(fk[k], w[3], acc3);
    }
    float* out = im + (size_t)edge_dst[e] * 128 + o0;
    atomicAdd(out + 0, acc0);
    atomicAdd(out + 1, acc1);
    atomicAdd(out + 2, acc2);
    atomicAdd(out + 3, acc3);
}

__global__ void k_relu(const float* __restrict__ in, float* __restrict__ out, int n4) {
    int t = blockIdx.x * blockDim.x + threadIdx.x;
    if (t >= n4) return;
    float4 v = ((const float4*)in)[t];
    v.x = fmaxf(v.x, 0.f); v.y = fmaxf(v.y, 0.f);
    v.z = fmaxf(v.z, 0.f); v.w = fmaxf(v.w, 0.f);
    ((float4*)out)[t] = v;
}

// agg[dst] += message[src] per edge; one wave/edge, float4/lane.
// message (25.6 MB) is L2-resident (192 MB L2): gather reads and scatter
// atomics are L2 traffic, not HBM.
__global__ void k_gather_scatter(const float* __restrict__ msg,
                                 const int* __restrict__ esrc,
                                 const int* __restrict__ edst,
                                 float* __restrict__ agg, int nEdges) {
    int t = blockIdx.x * blockDim.x + threadIdx.x;
    int e = t >> 5, lane = t & 31;
    if (e >= nEdges) return;
    int s = esrc[e], d = edst[e];
    float4 v = ((const float4*)(msg + (size_t)s * 128))[lane];
    float* ap = agg + (size_t)d * 128 + lane * 4;
    atomicAdd(ap + 0, v.x);
    atomicAdd(ap + 1, v.y);
    atomicAdd(ap + 2, v.z);
    atomicAdd(ap + 3, v.w);
}

// out[graph_ids[n]] += message[n]; one wave/node, float4/lane.
__global__ void k_pool(const float* __restrict__ msg,
                       const int* __restrict__ gids,
                       float* __restrict__ out, int nNodes) {
    int t = blockIdx.x * blockDim.x + threadIdx.x;
    int n = t >> 5, lane = t & 31;
    if (n >= nNodes) return;
    int g = gids[n];
    float4 v = ((const float4*)(msg + (size_t)n * 128))[lane];
    float* op = out + (size_t)g * 128 + lane * 4;
    atomicAdd(op + 0, v.x);
    atomicAdd(op + 1, v.y);
    atomicAdd(op + 2, v.z);
    atomicAdd(op + 3, v.w);
}

// ---------------------------------------------------------------------------
extern "C" void kernel_launch(void* const* d_in, const int* in_sizes, int n_in,
                              void* d_out, int out_size, void* d_ws, size_t ws_size,
                              hipStream_t stream) {
    const float* node_feat = (const float*)d_in[0];
    const float* edge_feat = (const float*)d_in[1];
    const int*   edge_src  = (const int*)d_in[2];
    const int*   edge_dst  = (const int*)d_in[3];
    const int*   graph_ids = (const int*)d_in[4];
    const float* W_n2l     = (const float*)d_in[5];
    const float* W_e2l     = (const float*)d_in[6];
    const float* W_rec     = (const float*)d_in[7];
    float*       out       = (float*)d_out;

    const int nNodes = in_sizes[0] / 128;   // 50000
    const int nEdges = in_sizes[2];         // 1600000
    const size_t nd  = (size_t)nNodes * 128;

    float*  im   = (float*)d_ws;                    // input_message  [nd] f32
    float*  msg  = im + nd;                         // message        [nd] f32
    float*  agg  = msg + nd;                        // aggregation    [nd] f32
    __bf16* abf  = (__bf16*)(agg + nd);             // bf16 A staging [nd]
    __bf16* wnbf = abf + nd;                        // W_n2l bf16 [128*128]
    __bf16* wrbf = wnbf + 128 * 128;                // W_rec bf16 [128*128]

    const int rowTiles   = (nNodes + 15) / 16;                                // 3125
    const int gemmBlocks = (rowTiles + TILES_PER_WAVE - 1) / TILES_PER_WAVE;  // 625
    const int edgeBlocks = (int)(((size_t)nEdges * 32 + 255) / 256);
    const int nodeBlocks = (int)(((size_t)nNodes * 32 + 255) / 256);
    const int cvtBlocks  = (int)((nd / 4 + 255) / 256);
    const int wBlocks    = (128 * 128 / 4 + 255) / 256;

    // One-time weight conversions to bf16.
    k_cvt_bf16<<<wBlocks, 256, 0, stream>>>(W_n2l, wnbf, 128 * 128 / 4);
    k_cvt_bf16<<<wBlocks, 256, 0, stream>>>(W_rec, wrbf, 128 * 128 / 4);

    // input_message = node_feat @ W_n2l.T   (WMMA on bf16-staged A)
    k_cvt_bf16<<<cvtBlocks, 256, 0, stream>>>(node_feat, abf, (int)(nd / 4));
    k_gemm_wmma<false, false><<<gemmBlocks, 256, 0, stream>>>(abf, wnbf, nullptr, im, nNodes);

    // input_message += scatter_add(edge_feat @ W_e2l.T, edge_dst)
    k_edge_emb<<<edgeBlocks, 256, 0, stream>>>(edge_feat, edge_dst, W_e2l, im, nEdges);

    // message = relu(input_message)
    k_relu<<<cvtBlocks, 256, 0, stream>>>(im, msg, (int)(nd / 4));

    for (int s = 0; s < 3; ++s) {
        hipMemsetAsync(agg, 0, nd * sizeof(float), stream);
        k_gather_scatter<<<edgeBlocks, 256, 0, stream>>>(msg, edge_src, edge_dst, agg, nEdges);
        // message = relu(input_message + agg @ W_rec.T)   (WMMA, fused epilogue)
        k_cvt_bf16<<<cvtBlocks, 256, 0, stream>>>(agg, abf, (int)(nd / 4));
        k_gemm_wmma<true, true><<<gemmBlocks, 256, 0, stream>>>(abf, wrbf, im, msg, nNodes);
    }

    hipMemsetAsync(out, 0, (size_t)N_GRAPHS * 128 * sizeof(float), stream);
    k_pool<<<nodeBlocks, 256, 0, stream>>>(msg, graph_ids, out, nNodes);
}
